// RecurrentTemporalDeformableDecoder_2448131359487
// MI455X (gfx1250) — matrix operand order repaired and checked by reading, not compiled, not run on hardware
//
#include <hip/hip_runtime.h>
#include <hip/hip_bf16.h>

typedef __bf16 v16bf __attribute__((ext_vector_type(16)));
typedef float  v8f   __attribute__((ext_vector_type(8)));
typedef int    v4i   __attribute__((ext_vector_type(4)));

#if __has_builtin(__builtin_amdgcn_global_load_async_to_lds_b128)
#define ASYNC_LDS 1
typedef __attribute__((address_space(1))) v4i gv4i;   // global (AS1) 4xi32
typedef __attribute__((address_space(3))) v4i lv4i;   // LDS    (AS3) 4xi32
#endif

constexpr int B_ = 4, T_ = 16, Q_ = 16, D_ = 256;
constexpr int NH_ = 8, NP_ = 4, K_ = 2, NL_ = 5;
constexpr int GH_ = 32, GW_ = 32, HW_ = 1024;
constexpr int HD_ = D_ / NH_;          // 32
constexpr int FFN_ = 4 * D_;           // 1024
constexpr int MROWS_ = B_ * T_ * HW_;  // 65536
constexpr float FPS_ = 30.0f;

// ---------------- workspace layout (bytes) ----------------
constexpr size_t OFF_POS2D = 0;
constexpr size_t OFF_ABSV  = OFF_POS2D + sizeof(float) * HW_ * D_;
constexpr size_t OFF_RELB  = OFF_ABSV  + sizeof(float) * T_ * D_;
constexpr size_t OFF_RBP   = OFF_RELB  + sizeof(float) * D_;
constexpr size_t OFF_PREV  = OFF_RBP   + sizeof(float) * 3 * D_;
constexpr size_t OFF_QBUF  = OFF_PREV  + sizeof(float) * B_ * Q_ * D_;
constexpr size_t OFF_REF   = OFF_QBUF  + sizeof(float) * B_ * Q_ * D_;
constexpr size_t OFF_OFFS  = OFF_REF   + sizeof(float) * B_ * Q_ * 2;
constexpr size_t OFF_AW    = OFF_OFFS  + sizeof(float) * B_ * Q_ * NH_ * NL_ * NP_ * 2;
constexpr size_t OFF_ATT   = OFF_AW    + sizeof(float) * B_ * Q_ * NH_ * NL_ * NP_;
constexpr size_t OFF_VP    = ((OFF_ATT + sizeof(float) * B_ * Q_ * D_) + 255) & ~size_t(255);
// Vp: 3 layers x MROWS_ x D_ bf16  (~96 MB)

// ---------------- helpers ----------------
__device__ __forceinline__ float block_reduce_sum(float v, float* red) {
  int tid = threadIdx.x;
  red[tid] = v;
  __syncthreads();
  for (int s = 128; s > 0; s >>= 1) {
    if (tid < s) red[tid] += red[tid + s];
    __syncthreads();
  }
  float r = red[0];
  __syncthreads();
  return r;
}

__device__ __forceinline__ float gelu_exact(float x) {
  return 0.5f * x * (1.0f + erff(x * 0.70710678118654752f));
}

// ---------------- precompute kernels ----------------
__global__ __launch_bounds__(256) void k_pos2d(float* __restrict__ pos2d) {
  int hw = blockIdx.x, d = threadIdx.x;
  float y = (float)(hw / GW_), x = (float)(hw % GW_);
  int group = d >> 6, j = d & 63;
  float omega = powf(10000.0f, -(float)j / 64.0f);
  float v;
  if (group == 0)      v = sinf(y * omega);
  else if (group == 1) v = cosf(y * omega);
  else if (group == 2) v = sinf(x * omega);
  else                 v = cosf(x * omega);
  pos2d[hw * D_ + d] = v;
}

__device__ __forceinline__ float time_pe_elem(float t, int d) {
  // half = 128; freqs[i] = exp(-log(1e4)*i/128)
  if (d < 128) return sinf(t * expf(-logf(10000.0f) * (float)d / 128.0f));
  return cosf(t * expf(-logf(10000.0f) * (float)(d - 128) / 128.0f));
}

__global__ __launch_bounds__(256) void k_absvec(const float* __restrict__ Wabs,
                                                const float* __restrict__ babs,
                                                float* __restrict__ absv) {
  __shared__ float pe[D_];
  int t = blockIdx.x, d = threadIdx.x;
  pe[d] = time_pe_elem((float)t / FPS_, d);
  __syncthreads();
  float s = babs[d];
  for (int k = 0; k < D_; ++k) s += pe[k] * Wabs[k * D_ + d];
  absv[t * D_ + d] = s;
}

__global__ __launch_bounds__(256) void k_relbias(const float* __restrict__ Wrel,
                                                 const float* __restrict__ brel,
                                                 float* __restrict__ relb) {
  __shared__ float pe[D_];
  int d = threadIdx.x;
  float acc = 0.0f;
  for (int l = 0; l < NL_; ++l) {
    float dt = (float)(l - K_) / FPS_;
    __syncthreads();
    pe[d] = time_pe_elem(dt, d);
    __syncthreads();
    float s = 0.0f;
    for (int k = 0; k < D_; ++k) s += pe[k] * Wrel[k * D_ + d];
    acc += s;
  }
  relb[d] = acc / (float)NL_ + brel[d];
}

__global__ __launch_bounds__(256) void k_rbp(const float* __restrict__ relb,
                                             const float* __restrict__ W,
                                             const float* __restrict__ b,
                                             float* __restrict__ rbp_out) {
  __shared__ float rb[D_];
  int d = threadIdx.x;
  rb[d] = relb[d];
  __syncthreads();
  float s = b[d];
  for (int k = 0; k < D_; ++k) s += rb[k] * W[k * D_ + d];
  rbp_out[d] = s;
}

__global__ __launch_bounds__(256) void k_init_prev(const float* __restrict__ initq,
                                                   float* __restrict__ prev) {
  int row = blockIdx.x, d = threadIdx.x;       // row = b*Q + q
  int q = row % Q_;
  prev[row * D_ + d] = initq[q * D_ + d];
}

// ---------------- WMMA GEMM: Vp = (patch + pos2d + abs) @ Wv + bv ----------------
__global__ __launch_bounds__(256) void vproj_gemm(const float* __restrict__ patch,
                                                  const float* __restrict__ pos2d,
                                                  const float* __restrict__ absv,
                                                  const float* __restrict__ Wv,
                                                  const float* __restrict__ bv,
                                                  __bf16* __restrict__ Vp) {
  __shared__ __bf16 As[64][32];    // [m][k] bf16, 4 KB
#ifdef ASYNC_LDS
  __shared__ float Bf[32][256];    // [k][n] raw fp32 weight tile, 32 KB (async-filled)
#else
  __shared__ __bf16 Bs[256][32];   // [n][k] bf16, 16 KB
#endif
  int tid  = threadIdx.x;
  int mbase = blockIdx.x * 64;
  int wave = tid >> 5, lane = tid & 31;
  int m0 = (wave & 3) * 16;        // wave's M sub-tile within block
  int n0 = (wave >> 2) * 128;      // wave's N half

  v8f acc[8];
#pragma unroll
  for (int i = 0; i < 8; ++i)
#pragma unroll
    for (int j = 0; j < 8; ++j) acc[i][j] = 0.0f;

  // A staging assignment: thread -> (row, 8 contiguous k)
  int arow  = tid >> 2;            // 0..63
  int acol0 = (tid & 3) * 8;       // 0,8,16,24
  int gm  = mbase + arow;
  int rem = gm % (T_ * HW_);
  int tt  = rem / HW_;
  int hw  = rem % HW_;
  const float* prow = patch + (size_t)gm * D_;
  const float* posr = pos2d + (size_t)hw * D_;
  const float* absr = absv + (size_t)tt * D_;

  for (int k0 = 0; k0 < D_; k0 += 32) {
    // ---- stage A (3-stream sum, fp32 -> bf16) ----
#pragma unroll
    for (int c = 0; c < 8; ++c) {
      int k = k0 + acol0 + c;
      As[arow][acol0 + c] = (__bf16)(prow[k] + posr[k] + absr[k]);
    }
    // ---- stage B (weight tile) ----
#ifdef ASYNC_LDS
    {
      // async copy 32x256 fp32 tile: 8 x b128 per thread slot (ASYNCcnt-tracked)
      const float* gsrc = Wv + (size_t)k0 * D_;
      float* ldst = &Bf[0][0];
#pragma unroll
      for (int i = 0; i < 8; ++i) {
        int o = i * 1024 + tid * 4;
        __builtin_amdgcn_global_load_async_to_lds_b128(
            (gv4i*)(gsrc + o), (lv4i*)(ldst + o), 0, 0);
      }
#if __has_builtin(__builtin_amdgcn_s_wait_asynccnt)
      __builtin_amdgcn_s_wait_asynccnt(0);
#else
      asm volatile("s_wait_asynccnt 0x0" ::: "memory");
#endif
    }
#else
#pragma unroll
    for (int k = 0; k < 32; ++k)
      Bs[tid][k] = (__bf16)Wv[(k0 + k) * D_ + tid];
#endif
    if (k0 + 32 < D_) __builtin_prefetch(prow + k0 + 32, 0, 1);  // global_prefetch_b8
    __syncthreads();

    // ---- A fragment: 16x32 bf16, contiguous k = 16*(lane>>4) + e ----
    v16bf af;
    {
      int r  = m0 + (lane & 15);
      int kb = (lane >> 4) * 16;
#pragma unroll
      for (int e = 0; e < 8; ++e) {
        af[e]     = As[r][kb + e];
        af[e + 8] = As[r][kb + 8 + e];
      }
    }
    // ---- B fragments + WMMA ----
#pragma unroll
    for (int nt = 0; nt < 8; ++nt) {
      int col = n0 + nt * 16 + (lane & 15);
      int kb  = (lane >> 4) * 16;
      v16bf bf;
#pragma unroll
      for (int e = 0; e < 16; ++e) {
#ifdef ASYNC_LDS
        bf[e] = (__bf16)Bf[kb + e][col];
#else
        bf[e] = Bs[col][kb + e];
#endif
      }
      acc[nt] = __builtin_amdgcn_wmma_f32_16x16x32_bf16(
          false, af, false, bf, (short)0, acc[nt], false, false);
    }
    __syncthreads();
  }

  // epilogue: C/D layout -> VGPR r: M = r + 8*(lane>=16), N = lane&15
  int mr = m0 + (lane >> 4) * 8;
  int nc = lane & 15;
#pragma unroll
  for (int nt = 0; nt < 8; ++nt) {
    int gn = n0 + nt * 16 + nc;
    float bias = bv[gn];
#pragma unroll
    for (int r = 0; r < 8; ++r) {
      int gmr = mbase + mr + r;
      Vp[(size_t)gmr * D_ + gn] = (__bf16)(acc[nt][r] + bias);
    }
  }
}

// ---------------- per-timestep kernels ----------------
__global__ __launch_bounds__(256) void k_merge(
    const float* __restrict__ temporal, const float* __restrict__ absv,
    const float* __restrict__ prev, const float* __restrict__ initq,
    const float* __restrict__ Wctx, const float* __restrict__ bctx,
    const float* __restrict__ Wmctx, const float* __restrict__ bmctx,
    const float* __restrict__ Wtime, const float* __restrict__ btime,
    const float* __restrict__ Wprev, const float* __restrict__ bprev,
    const float* __restrict__ Winit, const float* __restrict__ binit,
    const float* __restrict__ gmn, const float* __restrict__ bmn,
    float* __restrict__ qbuf, int t) {
  __shared__ float ctxv[D_];
  __shared__ float red[D_];
  int b = blockIdx.x, d = threadIdx.x;

  float s = bctx[d];
  for (int k = 0; k < D_; ++k) s += temporal[(b * T_ + t) * D_ + k] * Wctx[k * D_ + d];
  ctxv[d] = s;
  __syncthreads();

  float cx = bmctx[d];
  for (int k = 0; k < D_; ++k) cx += ctxv[k] * Wmctx[k * D_ + d];
  float tm = btime[d];
  for (int k = 0; k < D_; ++k) tm += absv[t * D_ + k] * Wtime[k * D_ + d];

  for (int q = 0; q < Q_; ++q) {
    const float* pr = prev + (size_t)(b * Q_ + q) * D_;
    const float* iq = initq + (size_t)q * D_;
    float ap = block_reduce_sum(pr[d] * Wprev[d], red) + bprev[0];
    float ai = block_reduce_sum(iq[d] * Winit[d], red) + binit[0];
    float mx = fmaxf(ap, ai);
    float e0 = expf(ap - mx), e1 = expf(ai - mx);
    float inv = 1.0f / (e0 + e1);
    float xv = (e0 * inv) * pr[d] + (e1 * inv) * iq[d] + cx + tm;
    float mean = block_reduce_sum(xv, red) * (1.0f / D_);
    float dv = xv - mean;
    float var = block_reduce_sum(dv * dv, red) * (1.0f / D_);
    qbuf[(size_t)(b * Q_ + q) * D_ + d] = dv * rsqrtf(var + 1e-5f) * gmn[d] + bmn[d];
  }
}

__global__ __launch_bounds__(256) void k_qprep(
    const float* __restrict__ qbuf, const float* __restrict__ rbp,
    const float* __restrict__ g1, const float* __restrict__ bn1,
    const float* __restrict__ Wref, const float* __restrict__ bref,
    const float* __restrict__ Woff, const float* __restrict__ boff,
    const float* __restrict__ Waw, const float* __restrict__ baw,
    float* __restrict__ refb, float* __restrict__ offb, float* __restrict__ awb_out) {
  __shared__ float qu[D_];
  __shared__ float red[D_];
  __shared__ float awb[NH_ * NL_ * NP_];
  int row = blockIdx.x, d = threadIdx.x;

  float qv = qbuf[(size_t)row * D_ + d];
  float mean = block_reduce_sum(qv, red) * (1.0f / D_);
  float dv = qv - mean;
  float var = block_reduce_sum(dv * dv, red) * (1.0f / D_);
  float qn = dv * rsqrtf(var + 1e-5f) * g1[d] + bn1[d];
  qu[d] = qn + rbp[d];
  __syncthreads();

  // ref = sigmoid(qn @ Wref + bref)  (2 outputs)
  for (int j = 0; j < 2; ++j) {
    float r = block_reduce_sum(qn * Wref[d * 2 + j], red) + bref[j];
    if (d == 0) refb[row * 2 + j] = 1.0f / (1.0f + expf(-r));
  }
  // off: 320 outputs
  for (int o = d; o < NH_ * NL_ * NP_ * 2; o += D_) {
    float s = boff[o];
    for (int k = 0; k < D_; ++k) s += qu[k] * Woff[k * (NH_ * NL_ * NP_ * 2) + o];
    offb[(size_t)row * (NH_ * NL_ * NP_ * 2) + o] = s;
  }
  // aw logits: 160 outputs
  if (d < NH_ * NL_ * NP_) {
    float s = baw[d];
    for (int k = 0; k < D_; ++k) s += qu[k] * Waw[k * (NH_ * NL_ * NP_) + d];
    awb[d] = s;
  }
  __syncthreads();
  // softmax per head over NL*NP = 20
  if (d < NH_) {
    int base = d * (NL_ * NP_);
    float mx = awb[base];
    for (int i = 1; i < NL_ * NP_; ++i) mx = fmaxf(mx, awb[base + i]);
    float sum = 0.0f;
    float e[NL_ * NP_];
    for (int i = 0; i < NL_ * NP_; ++i) { e[i] = expf(awb[base + i] - mx); sum += e[i]; }
    float inv = 1.0f / sum;
    for (int i = 0; i < NL_ * NP_; ++i)
      awb_out[(size_t)row * (NH_ * NL_ * NP_) + base + i] = e[i] * inv;
  }
}

__global__ __launch_bounds__(256) void k_msda(
    const __bf16* __restrict__ Vp, const float* __restrict__ refb,
    const float* __restrict__ offb, const float* __restrict__ awb,
    float* __restrict__ attn0, int t) {
  int row = blockIdx.x;            // b*Q + q
  int b = row / Q_;
  int h = threadIdx.x >> 5;        // head
  int c = threadIdx.x & 31;        // channel within head
  float rx = refb[row * 2 + 0];
  float ry = refb[row * 2 + 1];
  float acc = 0.0f;
  for (int l = 0; l < NL_; ++l) {
    int frame = t + l - K_;
    if (frame < 0 || frame >= T_) continue;   // uniform per block
    const __bf16* vbase = Vp + ((size_t)(b * T_ + frame) * HW_) * D_ + h * HD_ + c;
    for (int p = 0; p < NP_; ++p) {
      int oidx = ((h * NL_ + l) * NP_ + p) * 2;
      float ox = offb[(size_t)row * (NH_ * NL_ * NP_ * 2) + oidx];
      float oy = offb[(size_t)row * (NH_ * NL_ * NP_ * 2) + oidx + 1];
      float a  = awb[(size_t)row * (NH_ * NL_ * NP_) + h * (NL_ * NP_) + l * NP_ + p];
      float px = rx * GW_ + ox - 0.5f;
      float py = ry * GH_ + oy - 0.5f;
      float x0f = floorf(px), y0f = floorf(py);
      float wx = px - x0f, wy = py - y0f;
      int x0 = (int)x0f, y0 = (int)y0f;
      auto samp = [&](int xi, int yi) -> float {
        if (xi < 0 || xi >= GW_ || yi < 0 || yi >= GH_) return 0.0f;
        return (float)vbase[(size_t)(yi * GW_ + xi) * D_];
      };
      acc += a * (samp(x0, y0) * (1.0f - wx) * (1.0f - wy)
                + samp(x0 + 1, y0) * wx * (1.0f - wy)
                + samp(x0, y0 + 1) * (1.0f - wx) * wy
                + samp(x0 + 1, y0 + 1) * wx * wy);
    }
  }
  attn0[(size_t)row * D_ + h * HD_ + c] = acc;
}

__global__ __launch_bounds__(256) void k_post(
    float* __restrict__ qbuf, const float* __restrict__ attn0,
    const float* __restrict__ Wo, const float* __restrict__ bo,
    const float* __restrict__ Wp, const float* __restrict__ bp,
    const float* __restrict__ g2, const float* __restrict__ bn2,
    const float* __restrict__ W1, const float* __restrict__ b1,
    const float* __restrict__ W2, const float* __restrict__ b2) {
  __shared__ float arow[D_];
  __shared__ float attn[D_];
  __shared__ float hrow[D_];
  __shared__ float frow[FFN_];
  __shared__ float red[D_];
  int row = blockIdx.x, d = threadIdx.x;

  arow[d] = attn0[(size_t)row * D_ + d];
  float qv = qbuf[(size_t)row * D_ + d];
  __syncthreads();

  float s = bo[d];
  for (int k = 0; k < D_; ++k) s += arow[k] * Wo[k * D_ + d];
  attn[d] = s;
  __syncthreads();

  s = bp[d];
  for (int k = 0; k < D_; ++k) s += attn[k] * Wp[k * D_ + d];
  float outv = qv + s;

  float mean = block_reduce_sum(outv, red) * (1.0f / D_);
  float dv = outv - mean;
  float var = block_reduce_sum(dv * dv, red) * (1.0f / D_);
  hrow[d] = dv * rsqrtf(var + 1e-5f) * g2[d] + bn2[d];
  __syncthreads();

  for (int j = d; j < FFN_; j += D_) {
    float f = b1[j];
    for (int k = 0; k < D_; ++k) f += hrow[k] * W1[k * FFN_ + j];
    frow[j] = gelu_exact(f);
  }
  __syncthreads();

  s = b2[d];
  for (int k = 0; k < FFN_; ++k) s += frow[k] * W2[k * D_ + d];
  qbuf[(size_t)row * D_ + d] = outv + s;
}

__global__ __launch_bounds__(256) void k_commit(const float* __restrict__ qbuf,
                                                float* __restrict__ prev,
                                                float* __restrict__ out, int t) {
  int row = blockIdx.x, d = threadIdx.x;
  int b = row / Q_, q = row % Q_;
  float v = qbuf[(size_t)row * D_ + d];
  prev[(size_t)row * D_ + d] = v;
  out[(((size_t)b * T_ + t) * Q_ + q) * D_ + d] = v;
}

// ---------------- host launch ----------------
extern "C" void kernel_launch(void* const* d_in, const int* in_sizes, int n_in,
                              void* d_out, int out_size, void* d_ws, size_t ws_size,
                              hipStream_t stream) {
  (void)in_sizes; (void)n_in; (void)out_size; (void)ws_size;
  const float* patch    = (const float*)d_in[0];
  const float* temporal = (const float*)d_in[1];
  const float* initq = (const float*)d_in[2];
  const float* Wctx = (const float*)d_in[3];  const float* bctx = (const float*)d_in[4];
  const float* Wabs = (const float*)d_in[5];  const float* babs = (const float*)d_in[6];
  const float* Wrel = (const float*)d_in[7];  const float* brel = (const float*)d_in[8];
  const float* Wprev = (const float*)d_in[9];  const float* bprev = (const float*)d_in[10];
  const float* Winit = (const float*)d_in[11]; const float* binit = (const float*)d_in[12];
  const float* Wmctx = (const float*)d_in[13]; const float* bmctx = (const float*)d_in[14];
  const float* Wtime = (const float*)d_in[15]; const float* btime = (const float*)d_in[16];
  const float* gmn = (const float*)d_in[17];   const float* bmn = (const float*)d_in[18];
  auto LW = [&](int l, int j) { return (const float*)d_in[19 + 22 * l + j]; };
  // per-layer leaf offsets: 0 Wrbp 1 brbp 2 Wref 3 bref 4 Woff 5 boff 6 Waw 7 baw
  //                         8 Wv 9 bv 10 Wo 11 bo 12 Wp 13 bp 14 g1 15 bn1 16 g2 17 bn2
  //                         18 W1 19 b1 20 W2 21 b2

  char* ws = (char*)d_ws;
  float*  pos2d = (float*)(ws + OFF_POS2D);
  float*  absv  = (float*)(ws + OFF_ABSV);
  float*  relb  = (float*)(ws + OFF_RELB);
  float*  rbp   = (float*)(ws + OFF_RBP);
  float*  prev  = (float*)(ws + OFF_PREV);
  float*  qbuf  = (float*)(ws + OFF_QBUF);
  float*  refb  = (float*)(ws + OFF_REF);
  float*  offb  = (float*)(ws + OFF_OFFS);
  float*  awb   = (float*)(ws + OFF_AW);
  float*  attn0 = (float*)(ws + OFF_ATT);
  __bf16* Vp    = (__bf16*)(ws + OFF_VP);
  const size_t vp_layer = (size_t)MROWS_ * D_;

  k_pos2d<<<HW_, 256, 0, stream>>>(pos2d);
  k_absvec<<<T_, 256, 0, stream>>>(Wabs, babs, absv);
  k_relbias<<<1, 256, 0, stream>>>(Wrel, brel, relb);
  for (int l = 0; l < 3; ++l)
    k_rbp<<<1, 256, 0, stream>>>(relb, LW(l, 0), LW(l, 1), rbp + l * D_);
  for (int l = 0; l < 3; ++l)
    vproj_gemm<<<MROWS_ / 64, 256, 0, stream>>>(patch, pos2d, absv, LW(l, 8), LW(l, 9),
                                                Vp + (size_t)l * vp_layer);
  k_init_prev<<<B_ * Q_, 256, 0, stream>>>(initq, prev);

  for (int t = 0; t < T_; ++t) {
    k_merge<<<B_, 256, 0, stream>>>(temporal, absv, prev, initq,
                                    Wctx, bctx, Wmctx, bmctx, Wtime, btime,
                                    Wprev, bprev, Winit, binit, gmn, bmn, qbuf, t);
    for (int l = 0; l < 3; ++l) {
      k_qprep<<<B_ * Q_, 256, 0, stream>>>(qbuf, rbp + l * D_,
                                           LW(l, 14), LW(l, 15), LW(l, 2), LW(l, 3),
                                           LW(l, 4), LW(l, 5), LW(l, 6), LW(l, 7),
                                           refb, offb, awb);
      k_msda<<<B_ * Q_, 256, 0, stream>>>(Vp + (size_t)l * vp_layer, refb, offb, awb,
                                          attn0, t);
      k_post<<<B_ * Q_, 256, 0, stream>>>(qbuf, attn0,
                                          LW(l, 10), LW(l, 11), LW(l, 12), LW(l, 13),
                                          LW(l, 16), LW(l, 17), LW(l, 18), LW(l, 19),
                                          LW(l, 20), LW(l, 21));
    }
    k_commit<<<B_ * Q_, 256, 0, stream>>>(qbuf, prev, (float*)d_out, t);
  }
}
